// equi_conv_21603685499528
// MI455X (gfx1250) — matrix operand order repaired and checked by reading, not compiled, not run on hardware
//
#include <hip/hip_runtime.h>
#include <hip/hip_bf16.h>

typedef __attribute__((ext_vector_type(16))) _Float16 v16h;
typedef __attribute__((ext_vector_type(4)))  _Float16 v4h;
typedef __attribute__((ext_vector_type(8)))  float    v8f;
typedef __attribute__((ext_vector_type(4)))  float    v4f;

#define BB 2
#define PP 8192
#define NNB 24
#define C1N 32
#define C2N 32
#define KKN 13
#define AAN 12
#define MDIM (C2N*AAN)    // 384
#define CADIM (C1N*AAN)   // 384
#define MT (MDIM/16)      // 24 M-tiles
#define KT (CADIM/32)     // 12 K-chunks of 32
#define PTILE 64
#define NT (PTILE/16)     // 4 N-tiles

// ---- workspace layout (bytes). Total ~31.3 MB. ----
#define WT_HALFS (KKN*MT*KT*32*16)            // 1,916,928 halves
#define WS_WT   0
#define WS_KPOS (WT_HALFS*2)                   // 3,833,856
#define WS_BIAS (WS_KPOS + C2N*KKN*3*4)        // +4,992
#define WS_MDIR (WS_BIAS + C2N*AAN*4)          // +1,536
#define WS_PW   (WS_MDIR + BB*PP*3*4)          // +196,608 ; pw: 27.3 MB

// Fold W/k_pos_w + idx_map + trace permutations into f16 weights stored in
// WMMA A-fragment order: [k][mtile][kchunk][lane][16 halves].
// A-frag 16-bit layout: lane L holds row M = L%16; K(L,h) = (h<8 ? 8*(L/16)+h
//                                                           : 16 + 8*(L/16) + h-8)
__global__ void __launch_bounds__(256) prep_wt_kernel(
    const float* __restrict__ W, const float* __restrict__ k_pos_w,
    const int* __restrict__ idx_map, const int* __restrict__ tr_kv,
    const int* __restrict__ tr_a, _Float16* __restrict__ wt) {
  int t = blockIdx.x * 256 + threadIdx.x;      // 13*24*12*32 = 119,808 threads
  int lane  = t & 31;
  int rest  = t >> 5;
  int kc    = rest % KT;
  int mtile = (rest / KT) % MT;
  int k     = rest / (KT * MT);
  int m = mtile * 16 + (lane & 15);
  int d = m / AAN, r = m % AAN;
  int base = (lane >> 4) * 8;
  v16h o;
#pragma unroll
  for (int h = 0; h < 16; ++h) {
    int Kv = (h < 8) ? (base + h) : (16 + base + (h - 8));
    int ca = kc * 32 + Kv;
    int c = ca / AAN, a = ca % AAN;
    int kv = tr_kv[r * KKN + k];               // trace_idxv_rot[r][k]
    int av = tr_a[r * AAN + a];                // trace_idx_rot[r][a]
    int im = idx_map[kv * AAN + av];
    float val = W[(d * C1N + c) * 36 + im] * k_pos_w[d * 36 + im];
    o[h] = (_Float16)val;
  }
  *(v16h*)(wt + (size_t)t * 16) = o;
}

__global__ void prep_kpos_kernel(const float* __restrict__ k_pos_w,
                                 const int* __restrict__ idx_map,
                                 const float* __restrict__ vs,
                                 float* __restrict__ kpos) {
  int t = blockIdx.x * 256 + threadIdx.x;
  if (t >= C2N * KKN) return;
  int d = t / KKN, k = t % KKN;
  float x = 0.f, y = 0.f, z = 0.f;
  for (int a = 0; a < AAN; ++a) {
    float w = k_pos_w[d * 36 + idx_map[k * AAN + a]];
    x += w * vs[a * 3 + 0]; y += w * vs[a * 3 + 1]; z += w * vs[a * 3 + 2];
  }
  float n = fmaxf(sqrtf(x * x + y * y + z * z), 1e-12f);
  kpos[t * 3 + 0] = x / n; kpos[t * 3 + 1] = y / n; kpos[t * 3 + 2] = z / n;
}

__global__ void prep_bias_kernel(const float* __restrict__ bias,
                                 const int* __restrict__ lvl,
                                 const int* __restrict__ tr_kv,
                                 float* __restrict__ beff) {
  int t = blockIdx.x * 256 + threadIdx.x;
  if (t >= C2N * AAN) return;
  int d = t / AAN, r = t % AAN;
  float s = 0.f;
  for (int k = 0; k < KKN; ++k) s += bias[d * 5 + lvl[tr_kv[r * KKN + k]]];
  beff[t] = s;
}

__global__ void __launch_bounds__(256) meandir_kernel(
    const int* __restrict__ nidx, const float* __restrict__ verts,
    float* __restrict__ mdir) {
  int t = blockIdx.x * 256 + threadIdx.x;      // b*P+p, 16384 threads
  int b = t / PP;
  float vx = verts[t * 3 + 0], vy = verts[t * 3 + 1], vz = verts[t * 3 + 2];
  float sx = 0.f, sy = 0.f, sz = 0.f;
  for (int n = 0; n < NNB; ++n) {
    int j = nidx[t * NNB + n];
    int q = b * PP + j;
    float dx = verts[q * 3 + 0] - vx;
    float dy = verts[q * 3 + 1] - vy;
    float dz = verts[q * 3 + 2] - vz;
    float nn = fmaxf(sqrtf(dx * dx + dy * dy + dz * dz), 1e-12f);
    sx += dx / nn; sy += dy / nn; sz += dz / nn;
  }
  const float inv = 1.f / (float)NNB;
  mdir[t * 3 + 0] = sx * inv; mdir[t * 3 + 1] = sy * inv; mdir[t * 3 + 2] = sz * inv;
}

// pw[b][k][d][p] = relu(mean_dir[b,p,:] . k_pos_ori[d,k,:])
__global__ void __launch_bounds__(256) pw_kernel(
    const float* __restrict__ mdir, const float* __restrict__ kpos,
    float* __restrict__ pw) {
  __shared__ float sk[C2N * KKN * 3];
  for (int i = threadIdx.x; i < C2N * KKN * 3; i += 256) sk[i] = kpos[i];
  __syncthreads();
  int t = blockIdx.x * 256 + threadIdx.x;      // b*P+p
  int b = t / PP, p = t % PP;
  float mx = mdir[t * 3 + 0], my = mdir[t * 3 + 1], mz = mdir[t * 3 + 2];
  for (int k = 0; k < KKN; ++k)
    for (int d = 0; d < C2N; ++d) {
      const float* kp = &sk[(d * KKN + k) * 3];
      float v = fmaxf(mx * kp[0] + my * kp[1] + mz * kp[2], 0.f);
      pw[((size_t)(b * KKN + k) * C2N + d) * PP + p] = v;
    }
}

// Fused: for each k, Agg_k[(d,r),p] = WT_k x F_k via WMMA, then
// out[(d,r),p] += pw[d,p,k] * Agg_k ; epilogue adds bias_eff + ReLU.
// All hot-loop addresses are one running pointer + compile-time immediate
// offsets; A-fragments are double-buffered (prefetch distance 1 WMMA group).
__global__ void __launch_bounds__(256) main_gemm_kernel(
    const float* __restrict__ fm, const _Float16* __restrict__ wt,
    const float* __restrict__ pw, const float* __restrict__ beff,
    float* __restrict__ out) {
  // F_k tile staged in B-fragment order: [kchunk][ntile][lane] x v16h = 48KB
  __shared__ v16h ldsB[KT * NT * 32];
  int blk = blockIdx.x;
  int b  = blk / (PP / PTILE);
  int p0 = (blk % (PP / PTILE)) * PTILE;
  int lane = threadIdx.x & 31;
  int wave = threadIdx.x >> 5;

  // Running pointers: constant strides per k, immediate offsets inside.
  const _Float16* wlane = wt + ((size_t)(wave * 3) * KT * 32 + lane) * 16;
  const float*    pwl   = pw + ((size_t)b * KKN) * C2N * PP
                             + (size_t)(wave * 4) * PP + p0 + (lane & 15);

  v8f zero = {};
  v8f outacc[12];
#pragma unroll
  for (int i = 0; i < 12; ++i) outacc[i] = zero;

  for (int k = 0; k < KKN; ++k) {
    // ---- stage F_k (c,a)x(p) tile -> LDS f16 fragments ----
    // 4 consecutive elements (a-groups of 4) never cross a fragment row:
    // b128 coalesced global load + packed b64 LDS store.
    _Float16* lp = (_Float16*)ldsB;
#pragma unroll 4
    for (int j = 0; j < 24; ++j) {
      int e = (threadIdx.x + j * 256) * 4;     // 24576 elems, a fastest
      int a = e % AAN;                          // in {0,4,8}
      int p = (e / AAN) & 63;
      int c = e / (AAN * 64);
      v4f vv = *(const v4f*)(fm +
          ((((size_t)b * C1N + c) * KKN + k) * PP + p0 + p) * AAN + a);
      int ca = c * AAN + a;                     // multiple of 4
      int kt = ca >> 5, kk = ca & 31;
      int nt = p >> 4,  nn = p & 15;
      int lb = nn + (((kk >> 3) & 1) << 4);     // fragment lane row
      int hb = (kk & 7) + ((kk >> 4) << 3);     // multiple of 4
      v4h hv;
      hv[0] = (_Float16)vv[0]; hv[1] = (_Float16)vv[1];
      hv[2] = (_Float16)vv[2]; hv[3] = (_Float16)vv[3];
      *(v4h*)(&lp[((kt * NT + nt) * 32 + lb) * 16 + hb]) = hv;
    }
    __syncthreads();

#pragma unroll
    for (int i = 0; i < 3; ++i) {
      v8f acc[4];
#pragma unroll
      for (int nt = 0; nt < 4; ++nt) acc[nt] = zero;

      v16h afr[2];
      afr[0] = *(const v16h*)(wlane + (i * KT + 0) * 512);
#pragma unroll
      for (int kc = 0; kc < KT; ++kc) {
        if (kc + 1 < KT)                        // prefetch next A-fragment
          afr[(kc + 1) & 1] =
              *(const v16h*)(wlane + (i * KT + kc + 1) * 512);
#pragma unroll
        for (int nt = 0; nt < 4; ++nt) {
          v16h bfr = ldsB[(kc * NT + nt) * 32 + lane];
          acc[nt] = __builtin_amdgcn_wmma_f32_16x16x32_f16(
              false, afr[kc & 1], false, bfr, (short)0, acc[nt], false, false);
        }
      }

      // dA = 4*wave + i ; tile rows span dA (rows < 12-4i) and dA+1.
#pragma unroll
      for (int nt = 0; nt < 4; ++nt) {
        float pwA = pwl[(size_t)i * PP + nt * 16];
        float pwB = pwl[(size_t)(i + 1) * PP + nt * 16];
#pragma unroll
        for (int v = 0; v < 8; ++v) {
          const int lo1 = (4 * i + v     >= 12) ? 1 : 0;  // lanes 0-15
          const int hi1 = (4 * i + 8 + v >= 12) ? 1 : 0;  // lanes 16-31
          float pv;
          if (lo1 == hi1) pv = lo1 ? pwB : pwA;
          else            pv = (lane < 16) ? pwA : pwB;   // lo1=0, hi1=1
          outacc[i * 4 + nt][v] += acc[nt][v] * pv;
        }
      }
    }
    wlane += (size_t)MT * KT * 512;            // next k: +294,912 B
    pwl   += (size_t)C2N * PP;                 // next k: +1 MB
    __syncthreads();
  }

  // ---- epilogue: bias + relu, write out[b][d][p][r] ----
#pragma unroll
  for (int i = 0; i < 3; ++i) {
    int m0 = (wave * 3 + i) * 16;
#pragma unroll
    for (int nt = 0; nt < 4; ++nt) {
      int p = p0 + nt * 16 + (lane & 15);
#pragma unroll
      for (int v = 0; v < 8; ++v) {
        int m = m0 + ((lane >= 16) ? (8 + v) : v);
        int d = m / AAN, r = m % AAN;
        float o = outacc[i * 4 + nt][v] + beff[d * AAN + r];
        o = fmaxf(o, 0.f);
        out[(((size_t)b * C2N + d) * PP + p) * AAN + r] = o;
      }
    }
  }
}

extern "C" void kernel_launch(void* const* d_in, const int* in_sizes, int n_in,
                              void* d_out, int out_size, void* d_ws, size_t ws_size,
                              hipStream_t stream) {
  const int*   nidx    = (const int*)  d_in[0];   // neighbor_index (B,P,N)
  const float* verts   = (const float*)d_in[1];   // vertices (B,P,3)
  const float* fm      = (const float*)d_in[2];   // feature_map (B,C1,K,P,A)
  const float* W       = (const float*)d_in[3];   // (C2,C1,36)
  const float* bias    = (const float*)d_in[4];   // (C2,5)
  const float* kposw   = (const float*)d_in[5];   // (C2,36)
  const float* vs      = (const float*)d_in[6];   // (12,3)
  const int*   idx_map = (const int*)  d_in[7];   // (156,)
  const int*   tr_kv   = (const int*)  d_in[8];   // (A,K)
  const int*   tr_a    = (const int*)  d_in[9];   // (A,A)
  const int*   lvl     = (const int*)  d_in[10];  // (13,)
  (void)in_sizes; (void)n_in; (void)out_size; (void)ws_size;

  char* ws = (char*)d_ws;
  _Float16* wt  = (_Float16*)(ws + WS_WT);
  float*    kp  = (float*)(ws + WS_KPOS);
  float*    be  = (float*)(ws + WS_BIAS);
  float*    md  = (float*)(ws + WS_MDIR);
  float*    pwb = (float*)(ws + WS_PW);
  float*    out = (float*)d_out;

  prep_wt_kernel  <<<(KKN*MT*KT*32)/256, 256, 0, stream>>>(W, kposw, idx_map, tr_kv, tr_a, wt);
  prep_kpos_kernel<<<2, 256, 0, stream>>>(kposw, idx_map, vs, kp);
  prep_bias_kernel<<<2, 256, 0, stream>>>(bias, lvl, tr_kv, be);
  meandir_kernel  <<<(BB*PP)/256, 256, 0, stream>>>(nidx, verts, md);
  pw_kernel       <<<(BB*PP)/256, 256, 0, stream>>>(md, kp, pwb);
  main_gemm_kernel<<<BB*(PP/PTILE), 256, 0, stream>>>(fm, wt, pwb, be, out);
}
